// Enhancer_36043365548514
// MI455X (gfx1250) — matrix operand order repaired and checked by reading, not compiled, not run on hardware
//
#include <hip/hip_runtime.h>

// ---------------------------------------------------------------------------
// CDNA5 (gfx1250) sparse-conv network, bf16 WMMA with f32 accumulation.
// Core primitive: out[N,Cout] = sum_k gather(feats, nbr[k]) @ W[k] + b
// One wave32 per 16-row tile; CT (1/2/4) 16-wide column tiles per wave so
// every gathered A fragment feeds CT v_wmma_f32_16x16x32_bf16 issues.
// CIN is a template parameter => K-chunk loop fully unrolled, guards folded.
// ---------------------------------------------------------------------------

typedef __attribute__((ext_vector_type(16))) __bf16 v16bf;
typedef __attribute__((ext_vector_type(8)))  float  v8f;

union Frag {             // 32 bytes = one 16x32-bf16 fragment per lane
    v16bf  v;
    uint4  q[2];
    __bf16 h[16];
};

// ---------------------------------------------------------------------------
// Generic gather-conv (sconv3 with nbr, or sconv1 with nbr==nullptr, Kk=1).
//   feats : [N, CIN] bf16 (row major)
//   Wt    : [Kk, Cout, CIN] bf16 (pre-transposed: B-fragment K contiguous)
//   out   : written at out[row*ldout + coff + col]; optional bf16 and/or f32
//   resv  : optional residual (same indexing), added after bias/relu
// ---------------------------------------------------------------------------
template<int CIN, int CT>
__global__ __launch_bounds__(32)
void sconv_wmma_kernel(const __bf16* __restrict__ feats,
                       const __bf16* __restrict__ Wt,
                       const float*  __restrict__ bias,
                       const int*    __restrict__ nbr,     // [Kk, N] or null
                       __bf16*       __restrict__ out_bf,  // may be null
                       float*        __restrict__ out_f32, // may be null
                       const __bf16* __restrict__ resv,    // may be null
                       int Npts, int Cout, int Kk,
                       int ldout, int coff, int relu)
{
    const int lane  = threadIdx.x & 31;
    const int half  = lane >> 4;      // 0: lanes 0-15, 1: lanes 16-31
    const int m     = lane & 15;
    const int row0  = blockIdx.x * 16;
    const int cbase = blockIdx.y * (16 * CT);
    const int row   = row0 + m;       // point row this lane gathers for A

    const bool rowvalid = (row < Npts);

    v8f acc[CT] = {};

    for (int k = 0; k < Kk; ++k) {
        int gidx;
        if (nbr) gidx = rowvalid ? nbr[(size_t)k * Npts + row] : -1;
        else     gidx = rowvalid ? row : -1;
        const bool rowok = (gidx >= 0);
        const __bf16* arow = feats + (size_t)(rowok ? gidx : 0) * CIN;

#pragma unroll
        for (int cin0 = 0; cin0 < CIN; cin0 += 32) {
            // ---- A fragment (16x32 bf16): per-lane layout from ISA 7.12.2
            //   elements 0..7  at K = cin0 + 8*half
            //   elements 8..15 at K = cin0 + 16 + 8*half
            Frag a;
            const int lo = cin0 + 8 * half;
            const int hi = cin0 + 16 + 8 * half;
            if (rowok && (lo + 8) <= CIN) {
                a.q[0] = *(const uint4*)(arow + lo);
            } else {
#pragma unroll
                for (int e = 0; e < 8; ++e) {
                    const int kk = lo + e;
                    a.h[e] = (rowok && kk < CIN) ? arow[kk] : (__bf16)0.0f;
                }
            }
            if (rowok && (hi + 8) <= CIN) {
                a.q[1] = *(const uint4*)(arow + hi);
            } else {
#pragma unroll
                for (int e = 0; e < 8; ++e) {
                    const int kk = hi + e;
                    a.h[8 + e] = (rowok && kk < CIN) ? arow[kk] : (__bf16)0.0f;
                }
            }

            // ---- CT x { B fragment + WMMA }, reusing the gathered A
            const int kb = hi;   // B: lane = column, same K order as A (lo..lo+7, hi..hi+7)
#pragma unroll
            for (int t = 0; t < CT; ++t) {
                const int  col   = cbase + t * 16 + m;
                const bool colok = (col < Cout);
                const __bf16* brow = Wt + ((size_t)k * Cout + (colok ? col : 0)) * CIN;
                Frag b;
                if (colok && (kb + 8) <= CIN) {
                    b.q[0] = *(const uint4*)(brow + lo);
                    b.q[1] = *(const uint4*)(brow + hi);
                } else {
#pragma unroll
                    for (int e = 0; e < 16; ++e) {
                        const int kk = (e < 8) ? (lo + e) : (hi + e - 8);
                        b.h[e] = (colok && kk < CIN) ? brow[kk] : (__bf16)0.0f;
                    }
                }
                // D = A x B + C, f32 accumulation (EXEC all-ones here)
                acc[t] = __builtin_amdgcn_wmma_f32_16x16x32_bf16(
                    false, a.v, false, b.v, (short)0, acc[t], false, false);
            }
        }
    }

    // ---- epilogue: bias, relu, residual, concat offset, dual-format store
#pragma unroll
    for (int t = 0; t < CT; ++t) {
        const int col = cbase + t * 16 + m;
        if (col < Cout) {
            const float bv = bias[col];
#pragma unroll
            for (int r = 0; r < 8; ++r) {
                const int orow = row0 + r + 8 * half;   // D layout: VGPR r -> row
                if (orow < Npts) {
                    float v = acc[t][r] + bv;
                    if (relu) v = fmaxf(v, 0.0f);
                    const size_t o = (size_t)orow * ldout + coff + col;
                    if (resv)    v += (float)resv[o];
                    if (out_bf)  out_bf[o]  = (__bf16)v;
                    if (out_f32) out_f32[o] = v;
                }
            }
        }
    }
}

// ---------------------------------------------------------------------------
// Cin==1 gather conv (conv0: 1->64 with relu) -> bf16 output. VALU path: WMMA
// with K=32 would waste 31/32 of the MACs here.
// ---------------------------------------------------------------------------
__global__ __launch_bounds__(256)
void conv_cin1_kernel(const float* __restrict__ x,      // [N,1]
                      const float* __restrict__ w,      // [K][Cout]
                      const float* __restrict__ b,      // [Cout]
                      const int*   __restrict__ nbr,    // [K,N]
                      __bf16*      __restrict__ out,    // [N,Cout]
                      int Npts, int Cout, int Kk)
{
    const int i = blockIdx.x * blockDim.x + threadIdx.x;
    if (i >= Npts) return;
    float g[27];
    for (int k = 0; k < Kk; ++k) {
        const int idx = nbr[(size_t)k * Npts + i];
        g[k] = (idx >= 0) ? x[idx] : 0.0f;
    }
    for (int c = 0; c < Cout; ++c) {
        float acc = b[c];
        for (int k = 0; k < Kk; ++k) acc = fmaf(g[k], w[k * Cout + c], acc);
        out[(size_t)i * Cout + c] = (__bf16)fmaxf(acc, 0.0f);
    }
}

// ---------------------------------------------------------------------------
// Final skip branch: d_out[i,0:3] += sconv3(x_feats(1ch), out1_w, out1_b)
// ---------------------------------------------------------------------------
__global__ __launch_bounds__(256)
void conv_cin1_add3_kernel(const float* __restrict__ x,
                           const float* __restrict__ w,   // [K][3]
                           const float* __restrict__ b,   // [3]
                           const int*   __restrict__ nbr,
                           float*       __restrict__ out, // [N,3] (accumulate)
                           int Npts, int Kk)
{
    const int i = blockIdx.x * blockDim.x + threadIdx.x;
    if (i >= Npts) return;
    float a0 = b[0], a1 = b[1], a2 = b[2];
    for (int k = 0; k < Kk; ++k) {
        const int idx = nbr[(size_t)k * Npts + i];
        const float g = (idx >= 0) ? x[idx] : 0.0f;
        a0 = fmaf(g, w[k * 3 + 0], a0);
        a1 = fmaf(g, w[k * 3 + 1], a1);
        a2 = fmaf(g, w[k * 3 + 2], a2);
    }
    out[(size_t)i * 3 + 0] += a0;
    out[(size_t)i * 3 + 1] += a1;
    out[(size_t)i * 3 + 2] += a2;
}

// ---------------------------------------------------------------------------
// Weight conversion: f32 [K][Cin][Cout] -> bf16 [K][Cout][Cin] (transposed so
// B-fragment K is contiguous).
// ---------------------------------------------------------------------------
__global__ __launch_bounds__(256)
void wconv_kernel(const float* __restrict__ w, __bf16* __restrict__ wt,
                  int Kk, int Cin, int Cout)
{
    const int tid   = blockIdx.x * blockDim.x + threadIdx.x;
    const int total = Kk * Cin * Cout;
    if (tid >= total) return;
    const int k   = tid / (Cin * Cout);
    const int rem = tid % (Cin * Cout);
    const int ci  = rem / Cout;
    const int co  = rem % Cout;
    wt[((size_t)k * Cout + co) * Cin + ci] = (__bf16)w[tid];
}

// ---------------------------------------------------------------------------
// Host orchestration
// ---------------------------------------------------------------------------
extern "C" void kernel_launch(void* const* d_in, const int* in_sizes, int n_in,
                              void* d_out, int out_size, void* d_ws, size_t ws_size,
                              hipStream_t stream)
{
    (void)n_in; (void)out_size; (void)ws_size;
    const int N = in_sizes[0];   // 150000 points
    const int K = 27;

    const float* x   = (const float*)d_in[0];
    const int*   nbr = (const int*)d_in[53];   // last input: nbr_idx [27, N]

    // ---- bump allocator over workspace ----
    char*  ws  = (char*)d_ws;
    size_t off = 0;
    auto alloc = [&](size_t bytes) -> void* {
        void* p = ws + off;
        off += (bytes + 255) & ~(size_t)255;
        return p;
    };

    __bf16* F0 = (__bf16*)alloc((size_t)N * 128 * sizeof(__bf16));
    __bf16* F1 = (__bf16*)alloc((size_t)N * 128 * sizeof(__bf16));
    __bf16* H0 = (__bf16*)alloc((size_t)N * 32  * sizeof(__bf16));
    __bf16* H1 = (__bf16*)alloc((size_t)N * 32  * sizeof(__bf16));

    // convert one f32 weight to transposed bf16 in the arena
    auto convw = [&](int in_idx, int Kk, int Cin, int Cout) -> __bf16* {
        __bf16* wt = (__bf16*)alloc((size_t)Kk * Cin * Cout * sizeof(__bf16));
        const int total = Kk * Cin * Cout;
        wconv_kernel<<<(total + 255) / 256, 256, 0, stream>>>(
            (const float*)d_in[in_idx], wt, Kk, Cin, Cout);
        return wt;
    };

    // dispatch over compile-time (CIN, CT): CT = column tiles per wave
    auto sconv = [&](const __bf16* fin, int Cin, const __bf16* wt,
                     const float* bias, int Cout, const int* nb, int Kk,
                     __bf16* obf, float* of32, const __bf16* resv,
                     int ldout, int coff, int relu) {
        const int ct = (Cout % 64 == 0) ? 4 : (Cout % 32 == 0) ? 2 : 1;
        const dim3 blk(32);
#define LAUNCH_SCONV(CIN_, CT_)                                               \
        sconv_wmma_kernel<CIN_, CT_>                                          \
            <<<dim3((N + 15) / 16, (Cout + 16 * CT_ - 1) / (16 * CT_)),       \
               blk, 0, stream>>>(fin, wt, bias, nb, obf, of32, resv,          \
                                 N, Cout, Kk, ldout, coff, relu)
        switch (Cin) {
        case 16:
            if (ct == 4) LAUNCH_SCONV(16, 4);
            else if (ct == 2) LAUNCH_SCONV(16, 2);
            else LAUNCH_SCONV(16, 1);
            break;
        case 32:
            if (ct == 4) LAUNCH_SCONV(32, 4);
            else if (ct == 2) LAUNCH_SCONV(32, 2);
            else LAUNCH_SCONV(32, 1);
            break;
        case 64:
            if (ct == 4) LAUNCH_SCONV(64, 4);
            else if (ct == 2) LAUNCH_SCONV(64, 2);
            else LAUNCH_SCONV(64, 1);
            break;
        default:
            if (ct == 4) LAUNCH_SCONV(128, 4);
            else if (ct == 2) LAUNCH_SCONV(128, 2);
            else LAUNCH_SCONV(128, 1);
            break;
        }
#undef LAUNCH_SCONV
    };

    // ---- convert all matrix weights up front (in-order on stream) ----
    // input index map (dict insertion order): see reference setup_inputs()
    __bf16* W_r0_c00 = convw(3,  1, 64, 16);
    __bf16* W_r0_c01 = convw(5, 27, 16, 16);
    __bf16* W_r0_c02 = convw(7,  1, 16, 32);
    __bf16* W_r0_c10 = convw(9, 27, 64, 16);
    __bf16* W_r0_c11 = convw(11,27, 16, 32);
    __bf16* W_conv1  = convw(13,27, 64, 128);
    __bf16* W_r1_c00 = convw(15, 1, 128, 32);
    __bf16* W_r1_c01 = convw(17,27, 32, 32);
    __bf16* W_r1_c02 = convw(19, 1, 32, 64);
    __bf16* W_r1_c10 = convw(21,27, 128, 32);
    __bf16* W_r1_c11 = convw(23,27, 32, 64);
    __bf16* W_conv2  = convw(25,27, 128, 128);
    __bf16* W_r2_c00 = convw(27, 1, 128, 32);
    __bf16* W_r2_c01 = convw(29,27, 32, 32);
    __bf16* W_r2_c02 = convw(31, 1, 32, 64);
    __bf16* W_r2_c10 = convw(33,27, 128, 32);
    __bf16* W_r2_c11 = convw(35,27, 32, 64);
    __bf16* W_conv3  = convw(37,27, 128, 64);
    __bf16* W_r3_c00 = convw(39, 1, 64, 16);
    __bf16* W_r3_c01 = convw(41,27, 16, 16);
    __bf16* W_r3_c02 = convw(43, 1, 16, 32);
    __bf16* W_r3_c10 = convw(45,27, 64, 16);
    __bf16* W_r3_c11 = convw(47,27, 16, 32);
    __bf16* W_out0   = convw(49,27, 64, 3);

    auto B = [&](int i) { return (const float*)d_in[i]; };

    // inception-residual block: X [N,C] -> Y [N,C]
    auto incres = [&](__bf16* Wc00, const float* b00, __bf16* Wc01, const float* b01,
                      __bf16* Wc02, const float* b02, __bf16* Wc10, const float* b10,
                      __bf16* Wc11, const float* b11, int C, __bf16* X, __bf16* Y) {
        const int Cq = C / 4, Ch = C / 2;
        // h  = relu(x @ c00)                       -> H0 [N,Cq]
        sconv(X,  C,  Wc00, b00, Cq, nullptr, 1, H0, nullptr, nullptr, Cq, 0, 1);
        // h2 = relu(sconv3(h, c01))                -> H1 [N,Cq]
        sconv(H0, Cq, Wc01, b01, Cq, nbr,    27, H1, nullptr, nullptr, Cq, 0, 1);
        // out0 = h2 @ c02  -> Y[:,0:Ch] (+x residual, fused concat)
        sconv(H1, Cq, Wc02, b02, Ch, nullptr, 1, Y, nullptr, X, C, 0, 0);
        // h1 = relu(sconv3(x, c10))                -> H0 [N,Cq]
        sconv(X,  C,  Wc10, b10, Cq, nbr,    27, H0, nullptr, nullptr, Cq, 0, 1);
        // out1 = sconv3(h1, c11) -> Y[:,Ch:C] (+x residual)
        sconv(H0, Cq, Wc11, b11, Ch, nbr,    27, Y, nullptr, X, C, Ch, 0);
    };

    // ---- network ----
    // conv0: 1 -> 64, relu (VALU; WMMA K=32 would be 31/32 idle)
    conv_cin1_kernel<<<(N + 255) / 256, 256, 0, stream>>>(
        x, B(1), B(2), nbr, F0, N, 64, K);

    incres(W_r0_c00, B(4), W_r0_c01, B(6), W_r0_c02, B(8),
           W_r0_c10, B(10), W_r0_c11, B(12), 64, F0, F1);

    sconv(F1, 64,  W_conv1, B(14), 128, nbr, 27, F0, nullptr, nullptr, 128, 0, 1);

    incres(W_r1_c00, B(16), W_r1_c01, B(18), W_r1_c02, B(20),
           W_r1_c10, B(22), W_r1_c11, B(24), 128, F0, F1);

    sconv(F1, 128, W_conv2, B(26), 128, nbr, 27, F0, nullptr, nullptr, 128, 0, 1);

    incres(W_r2_c00, B(28), W_r2_c01, B(30), W_r2_c02, B(32),
           W_r2_c10, B(34), W_r2_c11, B(36), 128, F0, F1);

    sconv(F1, 128, W_conv3, B(38), 64, nbr, 27, F0, nullptr, nullptr, 64, 0, 1);

    incres(W_r3_c00, B(40), W_r3_c01, B(42), W_r3_c02, B(44),
           W_r3_c10, B(46), W_r3_c11, B(48), 64, F0, F1);

    // out0: sconv3 64 -> 3, f32 straight into d_out
    sconv(F1, 64, W_out0, B(50), 3, nbr, 27, nullptr, (float*)d_out, nullptr, 3, 0, 0);

    // out1: sconv3 1 -> 3 on raw input, accumulated into d_out
    conv_cin1_add3_kernel<<<(N + 255) / 256, 256, 0, stream>>>(
        x, B(51), B(52), nbr, (float*)d_out, N, K);
}
